// MiMoV2FlashAttention_74019466380010
// MI455X (gfx1250) — matrix operand order repaired and checked by reading, not compiled, not run on hardware
//
#include <hip/hip_runtime.h>
#include <hip/hip_bf16.h>

// ---------------------------------------------------------------------------
// Types for CDNA5 WMMA (wave32): V_WMMA_F32_16X16X32_BF16
// A: 16x32 bf16 (8 VGPRs/lane), B: 32x16 bf16 (8 VGPRs/lane), C/D: 16x16 f32
// ---------------------------------------------------------------------------
typedef __attribute__((ext_vector_type(16))) __bf16 bf16x16;
typedef __attribute__((ext_vector_type(8)))  __bf16 bf16x8;
typedef __attribute__((ext_vector_type(8)))  float  f32x8;
typedef __attribute__((ext_vector_type(4)))  float  f32x4;

union BF16Frag { bf16x16 v; bf16x8 h[2]; };

// CDNA5 async global->LDS copy (ASYNCcnt-tracked), with safe fallback.
// Toolchain signature (from diagnostics): param 0 is
// 'int __attribute__((vector_size(16))) *' in flat address space.
typedef int v4i_vs __attribute__((__vector_size__(16)));

#if __has_builtin(__builtin_amdgcn_global_load_async_to_lds_b128)
#define HAVE_ASYNC_LDS 1
#else
#define HAVE_ASYNC_LDS 0
#endif

__device__ __forceinline__ void wait_async_lds() {
#if __has_builtin(__builtin_amdgcn_s_wait_asynccnt)
    __builtin_amdgcn_s_wait_asynccnt(0);
#elif HAVE_ASYNC_LDS
    asm volatile("s_wait_asynccnt 0x0" ::: "memory");
#endif
}

// Model constants
#define SEQ   4096
#define HID   4096
#define NH    64
#define NKV   8
#define DQK   192
#define DVV   128
#define RD_   64
#define WIN   128
#define NB    (SEQ / WIN)   // 32

// ===========================================================================
// Kernel 1: generic GEMM  C[M,N] = A[M,K] * B[K,N], f32 in -> bf16 WMMA -> f32
// Block tile 128x128, 4 waves (2x2), each wave 64x64 (4x4 WMMA tiles), K=32/step.
// Global loads are batched into registers and pipelined one K-tile ahead so
// HBM latency overlaps the 16 WMMAs of the current tile.
// ===========================================================================
__global__ __launch_bounds__(128) void gemm_bf16_kernel(
    const float* __restrict__ A, const float* __restrict__ B,
    float* __restrict__ C, int M, int N, int K)
{
    __shared__ __bf16 As[128][32];   // row-major (m,k), 8 KB
    __shared__ __bf16 Bs[128][32];   // transposed  (n,k), 8 KB

    const int tid   = threadIdx.x;        // 0..127
    const int lane  = tid & 31;
    const int wave  = tid >> 5;           // 0..3
    const int wm    = wave >> 1;          // wave row (0..1)
    const int wn    = wave & 1;           // wave col (0..1)
    const int lhalf = lane >> 4;          // 0: lanes 0-15, 1: lanes 16-31
    const int l15   = lane & 15;

    const int bm = blockIdx.y * 128;
    const int bn = blockIdx.x * 128;

    // staging register tiles (A: 8x float4, B: 8x float4 per thread)
    const int a_c4 = (tid & 7) * 4;       // A k-offset 0..28
    const int a_r0 = tid >> 3;            // A row base 0..15 (stride 16)
    const int b_r  = tid >> 2;            // B k-row 0..31
    const int b_c0 = (tid & 3) * 32;      // B n-chunk

    f32x4 aR[8], bR[8];

    auto load_tiles = [&](int kb) {
        const float* asrc = A + (size_t)(bm + a_r0) * K + kb + a_c4;
        for (int i = 0; i < 8; ++i)
            aR[i] = *(const f32x4*)(asrc + (size_t)i * 16 * K);
        const float* bsrc = B + (size_t)(kb + b_r) * N + bn + b_c0;
        for (int i = 0; i < 8; ++i)
            bR[i] = *(const f32x4*)(bsrc + i * 4);
    };
    auto store_tiles = [&]() {
        for (int i = 0; i < 8; ++i) {
            const int r = a_r0 + i * 16;
            As[r][a_c4 + 0] = (__bf16)aR[i].x;
            As[r][a_c4 + 1] = (__bf16)aR[i].y;
            As[r][a_c4 + 2] = (__bf16)aR[i].z;
            As[r][a_c4 + 3] = (__bf16)aR[i].w;
        }
        for (int i = 0; i < 8; ++i) {
            const int c = b_c0 + i * 4;
            Bs[c + 0][b_r] = (__bf16)bR[i].x;
            Bs[c + 1][b_r] = (__bf16)bR[i].y;
            Bs[c + 2][b_r] = (__bf16)bR[i].z;
            Bs[c + 3][b_r] = (__bf16)bR[i].w;
        }
    };

    f32x8 acc[4][4] = {};

    load_tiles(0);
    for (int kb = 0; kb < K; kb += 32) {
        store_tiles();                       // waits on in-flight loads here
        __syncthreads();
        if (kb + 32 < K) load_tiles(kb + 32); // overlap with WMMA below

        // ---- fragments (per ISA 16-bit A/B VGPR layouts) ----
        BF16Frag afr[4], bfr[4];
        for (int i = 0; i < 4; ++i) {
            const __bf16* p = &As[wm * 64 + i * 16 + l15][0];
            // A: lanes 0-15 hold K 0-7 / 16-23, lanes 16-31 hold K 8-15 / 24-31
            afr[i].h[0] = *(const bf16x8*)(p + (lhalf ? 8  : 0));
            afr[i].h[1] = *(const bf16x8*)(p + (lhalf ? 24 : 16));
        }
        for (int j = 0; j < 4; ++j) {
            const __bf16* p = &Bs[wn * 64 + j * 16 + l15][0];
            // B: lanes 0-15 hold K 0-7 / 16-23, lanes 16-31 hold K 8-15 / 24-31
            bfr[j].h[0] = *(const bf16x8*)(p + (lhalf ? 8  : 0));
            bfr[j].h[1] = *(const bf16x8*)(p + (lhalf ? 24 : 16));
        }
        for (int i = 0; i < 4; ++i)
            for (int j = 0; j < 4; ++j)
                acc[i][j] = __builtin_amdgcn_wmma_f32_16x16x32_bf16(
                    false, afr[i].v, false, bfr[j].v,
                    (short)0, acc[i][j], false, false);
        __syncthreads();                     // LDS free before next store
    }

    // ---- epilogue: C layout — VGPR r: lanes 0-15 row r, lanes 16-31 row r+8
    for (int i = 0; i < 4; ++i)
        for (int j = 0; j < 4; ++j) {
            const int col = bn + wn * 64 + j * 16 + l15;
            for (int r = 0; r < 8; ++r) {
                const int row = bm + wm * 64 + i * 16 + (lhalf ? r + 8 : r);
                C[(size_t)row * N + col] = acc[i][j][r];
            }
        }
}

// ===========================================================================
// Kernel 2: RoPE (first 64 dims) + f32 -> bf16 pack. One thread per element.
// ===========================================================================
__global__ __launch_bounds__(256) void rope_pack_kernel(
    const float* __restrict__ x, const float* __restrict__ cosT,
    const float* __restrict__ sinT, __bf16* __restrict__ out,
    int nhd /* heads*hd */, int hd, int do_rope)
{
    const long idx = (long)blockIdx.x * 256 + threadIdx.x;
    const int  s   = (int)(idx / nhd);
    const int  d   = (int)(idx % hd);
    float v = x[idx];
    if (do_rope && d < RD_) {
        const float c  = cosT[s * RD_ + d];
        const float sn = sinT[s * RD_ + d];
        v = (d < RD_ / 2) ? (v * c - x[idx + RD_ / 2] * sn)
                          : (v * c + x[idx - RD_ / 2] * sn);
    }
    out[idx] = (__bf16)v;
}

// ===========================================================================
// Kernel 3: block-local attention with sink, one workgroup per (block, head).
// 8 waves x 16 query rows. LDS (dynamic 160 KB):
//   Ks : 256 x 192 bf16 (96 KB)  keys for [(n-1)W, (n+1)W), async-DMA'd
//   Vt : 128 x 256 bf16 (64 KB)  V transposed (dv-major) for B-frag reads
//   Ps : 8 x (16 x 256) bf16 (64 KB), aliases Ks after logits barrier
// ===========================================================================
__global__ __launch_bounds__(256) void attn_kernel(
    const __bf16* __restrict__ Q,    // [S][NH][DQK]
    const __bf16* __restrict__ Kk,   // [S][NKV][DQK]
    const __bf16* __restrict__ Vv,   // [S][NKV][DVV]
    const float*  __restrict__ sink, // [NH]
    float* __restrict__ O)           // [S][NH][DVV]
{
    extern __shared__ char smem[];
    __bf16* Ks = (__bf16*)smem;                  // 96 KB
    __bf16* Vt = (__bf16*)(smem + 96 * 1024);    // 64 KB
    __bf16* Ps = (__bf16*)smem;                  // aliases Ks

    const int n     = blockIdx.x;     // 0..31 query block
    const int h     = blockIdx.y;     // 0..63 head
    const int hk    = h >> 3;         // kv head (G=8)
    const int tid   = threadIdx.x;
    const int lane  = tid & 31;
    const int wave  = tid >> 5;       // 0..7
    const int lhalf = lane >> 4;
    const int l15   = lane & 15;

    // ---- stage K rows (key pos (n-1)*128 + rk, rk = 0..255) ----
    {
        const int rk = tid;
        const int gpos = (n - 1) * WIN + rk;
        __bf16* dst = Ks + rk * DQK;
        if (gpos < 0) {
            for (int i = 0; i < DQK; ++i) dst[i] = (__bf16)0.0f;
        } else {
            const __bf16* src = Kk + ((size_t)gpos * NKV + hk) * DQK;
#if HAVE_ASYNC_LDS
            for (int i = 0; i < DQK; i += 8)
                __builtin_amdgcn_global_load_async_to_lds_b128(
                    (v4i_vs*)(src + i),
                    (__attribute__((address_space(3))) v4i_vs*)(dst + i), 0, 0);
#else
            for (int i = 0; i < DQK; i += 8)
                *(bf16x8*)(dst + i) = *(const bf16x8*)(src + i);
#endif
        }
    }
    // ---- stage V transposed: Vt[dv][rk] (batched loads, scatter stores) ----
    {
        const int rk = tid;
        const int gpos = (n - 1) * WIN + rk;
        if (gpos < 0) {
            for (int d = 0; d < DVV; ++d) Vt[d * 256 + rk] = (__bf16)0.0f;
        } else {
            const __bf16* src = Vv + ((size_t)gpos * NKV + hk) * DVV;
            bf16x8 vreg[16];
            for (int d8 = 0; d8 < 16; ++d8)
                vreg[d8] = *(const bf16x8*)(src + d8 * 8);
            for (int d8 = 0; d8 < 16; ++d8)
                for (int u = 0; u < 8; ++u)
                    Vt[(d8 * 8 + u) * 256 + rk] = vreg[d8][u];
        }
    }

    // ---- Q fragments straight from global (this wave's 16 rows) ----
    BF16Frag qfr[6];
    {
        const int qrow = n * WIN + wave * 16 + l15;   // A layout: row = l15
        const __bf16* qsrc = Q + ((size_t)qrow * NH + h) * DQK;
        for (int kk = 0; kk < 6; ++kk) {
            qfr[kk].h[0] = *(const bf16x8*)(qsrc + kk * 32 + (lhalf ? 8  : 0));
            qfr[kk].h[1] = *(const bf16x8*)(qsrc + kk * 32 + (lhalf ? 24 : 16));
        }
    }
    wait_async_lds();
    __syncthreads();

    // ---- logits: 16 rows x 256 keys, K = 192 (6 WMMA k-steps x 16 N-tiles)
    f32x8 acc[16] = {};
    for (int kk = 0; kk < 6; ++kk) {
        for (int t = 0; t < 16; ++t) {
            BF16Frag b;
            const __bf16* kp = Ks + (t * 16 + l15) * DQK + kk * 32;
            b.h[0] = *(const bf16x8*)(kp + (lhalf ? 8  : 0));
            b.h[1] = *(const bf16x8*)(kp + (lhalf ? 24 : 16));
            acc[t] = __builtin_amdgcn_wmma_f32_16x16x32_bf16(
                false, qfr[kk].v, false, b.v, (short)0, acc[t], false, false);
        }
    }

    // ---- masked softmax with sink (row halves live on 16 lanes) ----
    const float scale = 0.072168784f;  // 192^-0.5
    const float NEGF  = -3.0e38f;
    const float b_h   = sink[h];
    for (int r = 0; r < 8; ++r) {
        const int qi = wave * 16 + (lhalf ? r + 8 : r);   // row in block
        float mx = b_h;
        for (int t = 0; t < 16; ++t) {
            const int c = t * 16 + l15;                   // key col 0..255
            const bool ok = (c >= WIN) ? ((c - WIN) <= qi)
                                       : ((n > 0) && (c > qi));
            const float l = ok ? acc[t][r] * scale : NEGF;
            acc[t][r] = l;
            mx = fmaxf(mx, l);
        }
        for (int m = 1; m < 16; m <<= 1)
            mx = fmaxf(mx, __shfl_xor(mx, m, 32));
        float sum = 0.0f;
        for (int t = 0; t < 16; ++t) {
            const float p = __expf(acc[t][r] - mx);
            acc[t][r] = p;
            sum += p;
        }
        for (int m = 1; m < 16; m <<= 1)
            sum += __shfl_xor(sum, m, 32);
        sum += __expf(b_h - mx);                           // sink column
        const float inv = 1.0f / sum;
        for (int t = 0; t < 16; ++t) acc[t][r] *= inv;
    }

    // ---- all waves done reading Ks; stage P (aliases Ks) ----
    __syncthreads();
    __bf16* pw = Ps + wave * (16 * 256);
    for (int t = 0; t < 16; ++t)
        for (int r = 0; r < 8; ++r) {
            const int row = lhalf ? r + 8 : r;
            pw[row * 256 + t * 16 + l15] = (__bf16)acc[t][r];
        }
    __syncthreads();

    // ---- O = P(16x256) * V(256x128): 16 k-steps x 8 N-tiles ----
    f32x8 oacc[8] = {};
    for (int kk = 0; kk < 16; ++kk) {
        BF16Frag pf;
        const __bf16* pp = pw + l15 * 256 + kk * 32;
        pf.h[0] = *(const bf16x8*)(pp + (lhalf ? 8  : 0));
        pf.h[1] = *(const bf16x8*)(pp + (lhalf ? 24 : 16));
        for (int t = 0; t < 8; ++t) {
            BF16Frag vfr;
            const __bf16* vp = Vt + (t * 16 + l15) * 256 + kk * 32;
            vfr.h[0] = *(const bf16x8*)(vp + (lhalf ? 8  : 0));
            vfr.h[1] = *(const bf16x8*)(vp + (lhalf ? 24 : 16));
            oacc[t] = __builtin_amdgcn_wmma_f32_16x16x32_bf16(
                false, pf.v, false, vfr.v, (short)0, oacc[t], false, false);
        }
    }

    // ---- write attention output f32: O[s][h][dv] ----
    for (int t = 0; t < 8; ++t) {
        const int dv = t * 16 + l15;
        for (int r = 0; r < 8; ++r) {
            const int row = n * WIN + wave * 16 + (lhalf ? r + 8 : r);
            O[((size_t)row * NH + h) * DVV + dv] = oacc[t][r];
        }
    }
}

// ===========================================================================
// Launch: 3 projection GEMMs -> rope/pack -> attention -> output GEMM
// ===========================================================================
extern "C" void kernel_launch(void* const* d_in, const int* in_sizes, int n_in,
                              void* d_out, int out_size, void* d_ws, size_t ws_size,
                              hipStream_t stream) {
    const float* hidden = (const float*)d_in[0];
    // d_in[1] = position_ids (identity arange; position == seq index)
    const float* cosT   = (const float*)d_in[2];
    const float* sinT   = (const float*)d_in[3];
    const float* Wq     = (const float*)d_in[4];
    const float* Wk     = (const float*)d_in[5];
    const float* Wv     = (const float*)d_in[6];
    const float* Wo     = (const float*)d_in[7];
    const float* sinkB  = (const float*)d_in[8];
    float* out = (float*)d_out;

    char* ws = (char*)d_ws;
    size_t off = 0;
    float* qf = (float*)(ws + off);  off += (size_t)SEQ * NH  * DQK * 4;
    float* kf = (float*)(ws + off);  off += (size_t)SEQ * NKV * DQK * 4;
    float* vf = (float*)(ws + off);  off += (size_t)SEQ * NKV * DVV * 4;
    __bf16* Qb = (__bf16*)(ws + off); off += (size_t)SEQ * NH  * DQK * 2;
    __bf16* Kb = (__bf16*)(ws + off); off += (size_t)SEQ * NKV * DQK * 2;
    __bf16* Vb = (__bf16*)(ws + off); off += (size_t)SEQ * NKV * DVV * 2;
    float* attnO = (float*)(ws + off); off += (size_t)SEQ * NH * DVV * 4;

    // QKV projections (bf16 WMMA, f32 accumulate)
    gemm_bf16_kernel<<<dim3((NH * DQK) / 128, SEQ / 128), 128, 0, stream>>>(
        hidden, Wq, qf, SEQ, NH * DQK, HID);
    gemm_bf16_kernel<<<dim3((NKV * DQK) / 128, SEQ / 128), 128, 0, stream>>>(
        hidden, Wk, kf, SEQ, NKV * DQK, HID);
    gemm_bf16_kernel<<<dim3((NKV * DVV) / 128, SEQ / 128), 128, 0, stream>>>(
        hidden, Wv, vf, SEQ, NKV * DVV, HID);

    // RoPE + bf16 pack
    rope_pack_kernel<<<(SEQ * NH * DQK) / 256, 256, 0, stream>>>(
        qf, cosT, sinT, Qb, NH * DQK, DQK, 1);
    rope_pack_kernel<<<(SEQ * NKV * DQK) / 256, 256, 0, stream>>>(
        kf, cosT, sinT, Kb, NKV * DQK, DQK, 1);
    rope_pack_kernel<<<(SEQ * NKV * DVV) / 256, 256, 0, stream>>>(
        vf, cosT, sinT, Vb, NKV * DVV, DVV, 0);

    // Block-local attention with sink; 160 KB dynamic LDS (CDNA5: 320 KB/WGP)
    attn_kernel<<<dim3(NB, NH), 256, 160 * 1024, stream>>>(
        Qb, Kb, Vb, sinkB, attnO);

    // Output projection: (S x H*DV) @ Wo(H*DV x HID)
    gemm_bf16_kernel<<<dim3(HID / 128, SEQ / 128), 128, 0, stream>>>(
        attnO, Wo, out, SEQ, HID, NH * DVV);
}
